// MultiHeadAttention_49692771614843
// MI455X (gfx1250) — compile-verified, matrix operand order checked by reading, not run on hardware
//
#include <hip/hip_runtime.h>
#include <hip/hip_bf16.h>

typedef __attribute__((ext_vector_type(16))) __bf16 bf16x16;
typedef __attribute__((ext_vector_type(8)))  __bf16 bf16x8;
typedef __attribute__((ext_vector_type(4)))  __bf16 bf16x4;
typedef __attribute__((ext_vector_type(8)))  float  f32x8;

#define WMMA_BF16(A, Bm, C) \
  __builtin_amdgcn_wmma_f32_16x16x32_bf16(false, (A), false, (Bm), (short)0, (C), false, false)

static constexpr int B_  = 2;
static constexpr int T_  = 2048;
static constexpr int D_  = 1024;
static constexpr int H_  = 16;
static constexpr int DK_ = 64;
static constexpr int BT_ = B_ * T_;   // 4096

// ---------------------------------------------------------------------------
// Fragment loader (works for global or LDS pointers; inlined so addrspace is
// recovered and LDS uses ds_load_b128):
//   lane < 16 : k in {0..7} (elems 0..7) and {16..23} (elems 8..15)
//   lane >=16 : k in {8..15}            and {24..31}
// ---------------------------------------------------------------------------
__device__ inline bf16x16 load_frag(const __bf16* base, int row_stride) {
  const int lane = threadIdx.x & 31;
  const int r    = lane & 15;
  const int ko   = (lane & 16) ? 8 : 0;
  const __bf16* p = base + (size_t)r * row_stride + ko;
  union { bf16x16 v; bf16x8 h[2]; } u;
  u.h[0] = *(const bf16x8*)(p);
  u.h[1] = *(const bf16x8*)(p + 16);
  return u.v;
}

// Async global -> LDS copy of 16 bytes per lane (ASYNCcnt-tracked).
__device__ inline void async_load_b128(const __bf16* gptr, const __bf16* lptr) {
  unsigned ldsoff = (unsigned)(unsigned long long)lptr;  // LDS aperture: addr[31:0] = LDS offset
  asm volatile("global_load_async_to_lds_b128 %0, %1, off"
               :: "v"(ldsoff), "v"(gptr) : "memory");
}

// ---------------------------------------------------------------------------
// f32 -> bf16 conversion (x and weight matrices), 4 elems/thread
// ---------------------------------------------------------------------------
__global__ void cvt_f32_bf16_kernel(const float* __restrict__ in,
                                    __bf16* __restrict__ out, int n) {
  int i = (blockIdx.x * blockDim.x + threadIdx.x) * 4;
  if (i + 3 < n) {
    float4 f = *(const float4*)(in + i);
    bf16x4 o;
    o.x = (__bf16)f.x; o.y = (__bf16)f.y; o.z = (__bf16)f.z; o.w = (__bf16)f.w;
    *(bf16x4*)(out + i) = o;
  }
}

// ---------------------------------------------------------------------------
// Fused QKV projection + bias + RoPE (q,k) / transpose-store (v).
// One wave computes a 16(t) x 64(dk of one head) tile.
// ---------------------------------------------------------------------------
__global__ void qkv_rope_kernel(const __bf16* __restrict__ xb,
                                const __bf16* __restrict__ wqb,
                                const __bf16* __restrict__ wkb,
                                const __bf16* __restrict__ wvb,
                                const float* __restrict__ bq,
                                const float* __restrict__ bk,
                                const float* __restrict__ bv,
                                __bf16* __restrict__ Q,
                                __bf16* __restrict__ K,
                                __bf16* __restrict__ Vt) {
  const int wid  = blockIdx.x * (blockDim.x >> 5) + (threadIdx.x >> 5);
  const int proj = wid >> 12;          // 4096 waves per projection
  const int rem  = wid & 4095;
  const int mt   = rem >> 4;           // row tile 0..255
  const int h    = rem & 15;
  const int lane  = threadIdx.x & 31;
  const int half8 = (lane & 16) ? 8 : 0;

  const __bf16* w    = (proj == 0) ? wqb : (proj == 1) ? wkb : wvb;
  const float*  bias = (proj == 0) ? bq  : (proj == 1) ? bk  : bv;

  const int row0 = mt * 16;            // global row in [0, BT)
  f32x8 acc[4] = {};

  for (int k0 = 0; k0 < D_; k0 += 32) {
    bf16x16 a = load_frag(xb + (size_t)row0 * D_ + k0, D_);
#pragma unroll
    for (int ns = 0; ns < 4; ++ns) {
      bf16x16 bm = load_frag(w + (size_t)(h * DK_ + ns * 16) * D_ + k0, D_);
      acc[ns] = WMMA_BF16(a, bm, acc[ns]);
    }
  }

  const int b  = row0 / T_;
  const int tb = row0 % T_;

  if (proj < 2) {
    __bf16* dst = (proj == 0) ? Q : K;
#pragma unroll
    for (int ns = 0; ns < 4; ++ns) {
      const int   dk  = ns * 16 + (lane & 15);
      const float bs  = bias[h * DK_ + dk];
      const int   pi  = dk >> 1;
      const float inv_theta = __powf(10000.0f, -(float)(2 * pi) / 64.0f);
      const bool  odd = dk & 1;
#pragma unroll
      for (int g = 0; g < 8; ++g) {
        float v       = acc[ns][g] + bs;
        float partner = __shfl_xor(v, 1, 32);   // paired even/odd column
        float ang = (float)(tb + g + half8) * inv_theta;
        float sn, cs; __sincosf(ang, &sn, &cs);
        float r = odd ? (partner * sn + v * cs) : (v * cs - partner * sn);
        dst[((size_t)(b * H_ + h) * T_ + (tb + g + half8)) * DK_ + dk] = (__bf16)r;
      }
    }
  } else {  // V: store transposed [B,H,DK,T]
#pragma unroll
    for (int ns = 0; ns < 4; ++ns) {
      const int   dk = ns * 16 + (lane & 15);
      const float bs = bias[h * DK_ + dk];
#pragma unroll
      for (int g = 0; g < 8; ++g) {
        float v = acc[ns][g] + bs;
        Vt[((size_t)(b * H_ + h) * DK_ + dk) * T_ + (tb + g + half8)] = (__bf16)v;
      }
    }
  }
}

// ---------------------------------------------------------------------------
// Causal flash attention, transposed-score formulation, workgroup-cooperative:
// 8 waves of a block handle 8 consecutive 16-query tiles of one (b,h) and
// share each 32-key K/V chunk through LDS, staged with double-buffered
// global_load_async_to_lds_b128 (ASYNCcnt) so loads overlap compute.
// ---------------------------------------------------------------------------
__global__ void __launch_bounds__(256)
flash_attn_kernel(const __bf16* __restrict__ Q,
                  const __bf16* __restrict__ K,
                  const __bf16* __restrict__ Vt,
                  __bf16* __restrict__ ctxb) {
  __shared__ __align__(16) __bf16 shK[2][32 * DK_];  // 32 keys x 64 dk
  __shared__ __align__(16) __bf16 shV[2][DK_ * 32];  // 64 dk  x 32 keys

  const int bh   = blockIdx.x >> 4;       // (b,h)
  const int ttg  = blockIdx.x & 15;       // group of 8 query tiles
  const int wave = threadIdx.x >> 5;
  const int tt   = ttg * 8 + wave;        // this wave's 16-query tile
  const int lane  = threadIdx.x & 31;
  const int half8 = (lane & 16) ? 8 : 0;
  const int tcol  = tt * 16 + (lane & 15);

  const __bf16* Qb = Q  + ((size_t)bh * T_ + tt * 16) * DK_;
  const __bf16* Kb = K  + (size_t)bh * T_ * DK_;
  const __bf16* Vb = Vt + (size_t)bh * DK_ * T_;

  // cooperative-load mapping: 256 threads, one b128 each for K and for V
  const int tid = threadIdx.x;
  const int kr = tid >> 3, kq = tid & 7;  // K: 32 rows x 8 quads of 16B
  const int vr = tid >> 2, vq = tid & 3;  // V: 64 rows x 4 quads of 16B

  // Q as B-operand of S^T gemm (lane = query column)
  const bf16x16 q0 = load_frag(Qb, DK_);
  const bf16x16 q1 = load_frag(Qb + 32, DK_);

  f32x8 acc[4] = {};                      // ctx^T: 64 dk rows x 16 t cols
  float m = -3.0e38f, l = 0.0f;
  const int   t_hi   = tt * 16 + 15;
  const float scale  = 0.125f;            // 1/sqrt(64)
  const int   blk_hi = (ttg * 8 + 7) * 16 + 15;
  const int   nchunk = blk_hi / 32 + 1;

  // prologue: stage chunk 0 into buffer 0
  async_load_b128(Kb + (size_t)kr * DK_ + kq * 8, &shK[0][kr * DK_ + kq * 8]);
  async_load_b128(Vb + (size_t)vr * T_ + vq * 8,  &shV[0][vr * 32 + vq * 8]);

  for (int c = 0; c < nchunk; ++c) {
    const int s0  = c * 32;
    const int buf = c & 1;

    if (c + 1 < nchunk) {   // prefetch next chunk into other buffer
      const int sn = s0 + 32;
      async_load_b128(Kb + (size_t)(sn + kr) * DK_ + kq * 8,
                      &shK[buf ^ 1][kr * DK_ + kq * 8]);
      async_load_b128(Vb + (size_t)vr * T_ + sn + vq * 8,
                      &shV[buf ^ 1][vr * 32 + vq * 8]);
      asm volatile("s_wait_asynccnt 0x2" ::: "memory");  // chunk c complete
    } else {
      asm volatile("s_wait_asynccnt 0x0" ::: "memory");
    }
    __syncthreads();        // all waves' stages of chunk c visible

    if (s0 <= t_hi) {       // wave-uniform causal guard (EXEC stays all-ones)
      const __bf16* sk = &shK[buf][0];
      const __bf16* sv = &shV[buf][0];

      f32x8 st0 = {}, st1 = {};
      {
        bf16x16 a0 = load_frag(sk, DK_);
        bf16x16 a1 = load_frag(sk + 32, DK_);
        st0 = WMMA_BF16(a0, q0, st0);
        st0 = WMMA_BF16(a1, q1, st0);
        bf16x16 a2 = load_frag(sk + 16 * DK_, DK_);
        bf16x16 a3 = load_frag(sk + 16 * DK_ + 32, DK_);
        st1 = WMMA_BF16(a2, q0, st1);
        st1 = WMMA_BF16(a3, q1, st1);
      }

      // scale + causal mask + running max over keys (per query column)
      float s0v[8], s1v[8];
      float mloc = -3.0e38f;
#pragma unroll
      for (int g = 0; g < 8; ++g) {
        int sA = s0 + g + half8;
        int sB = sA + 16;
        float v0 = (sA > tcol) ? -1.0e9f : st0[g] * scale;
        float v1 = (sB > tcol) ? -1.0e9f : st1[g] * scale;
        s0v[g] = v0; s1v[g] = v1;
        mloc = fmaxf(mloc, fmaxf(v0, v1));
      }
      mloc = fmaxf(mloc, __shfl_xor(mloc, 16, 32));
      const float mnew  = fmaxf(m, mloc);
      const float alpha = __expf(m - mnew);

      // exp + in-lane repack into B-operand fragment for PV^T
      float lloc = 0.0f;
      bf16x16 pb;
#pragma unroll
      for (int g = 0; g < 8; ++g) {
        float p0 = __expf(s0v[g] - mnew);
        float p1 = __expf(s1v[g] - mnew);
        lloc += p0 + p1;
        pb[g]     = (__bf16)p0;   // elems 0..7  <-> s = s0 + g + 8*half
        pb[g + 8] = (__bf16)p1;   // elems 8..15 <-> s = s0+16 + g + 8*half
      }
      l = l * alpha + lloc + __shfl_xor(lloc, 16, 32);

      // ctx^T[d, t] += Vt[d, chunk] * P^T[chunk, t]
#pragma unroll
      for (int dt = 0; dt < 4; ++dt) {
        bf16x16 av = load_frag(sv + dt * 16 * 32, 32);
#pragma unroll
        for (int g = 0; g < 8; ++g) acc[dt][g] *= alpha;
        acc[dt] = WMMA_BF16(av, pb, acc[dt]);
      }
      m = mnew;
    }
    __syncthreads();        // everyone done reading buf before it is re-staged
  }

  const float invl = 1.0f / l;
  const int b = bh >> 4, h = bh & 15;
#pragma unroll
  for (int dt = 0; dt < 4; ++dt) {
#pragma unroll
    for (int g = 0; g < 8; ++g) {
      int d = h * DK_ + dt * 16 + g + half8;
      ctxb[(size_t)(b * T_ + tcol) * D_ + d] = (__bf16)(acc[dt][g] * invl);
    }
  }
}

// ---------------------------------------------------------------------------
// Output projection: out = ctx @ wo^T + bo  (f32 output)
// One wave computes a 16x64 tile.
// ---------------------------------------------------------------------------
__global__ void out_proj_kernel(const __bf16* __restrict__ ctxb,
                                const __bf16* __restrict__ wob,
                                const float* __restrict__ bo,
                                float* __restrict__ out) {
  const int wid = blockIdx.x * (blockDim.x >> 5) + (threadIdx.x >> 5);
  const int mt  = wid >> 4;       // 0..255
  const int nt  = wid & 15;       // 64-wide column tile
  const int lane  = threadIdx.x & 31;
  const int half8 = (lane & 16) ? 8 : 0;
  const int row0 = mt * 16;
  const int col0 = nt * 64;

  f32x8 acc[4] = {};
  for (int k0 = 0; k0 < D_; k0 += 32) {
    bf16x16 a = load_frag(ctxb + (size_t)row0 * D_ + k0, D_);
#pragma unroll
    for (int ns = 0; ns < 4; ++ns) {
      bf16x16 bm = load_frag(wob + (size_t)(col0 + ns * 16) * D_ + k0, D_);
      acc[ns] = WMMA_BF16(a, bm, acc[ns]);
    }
  }
#pragma unroll
  for (int ns = 0; ns < 4; ++ns) {
    const int   col = col0 + ns * 16 + (lane & 15);
    const float bs  = bo[col];
#pragma unroll
    for (int g = 0; g < 8; ++g)
      out[(size_t)(row0 + g + half8) * D_ + col] = acc[ns][g] + bs;
  }
}

// ---------------------------------------------------------------------------
extern "C" void kernel_launch(void* const* d_in, const int* in_sizes, int n_in,
                              void* d_out, int out_size, void* d_ws, size_t ws_size,
                              hipStream_t stream) {
  const float* x  = (const float*)d_in[0];
  const float* wq = (const float*)d_in[1];
  const float* bq = (const float*)d_in[2];
  const float* wk = (const float*)d_in[3];
  const float* bk = (const float*)d_in[4];
  const float* wv = (const float*)d_in[5];
  const float* bv = (const float*)d_in[6];
  const float* wo = (const float*)d_in[7];
  const float* bo = (const float*)d_in[8];
  float* out = (float*)d_out;

  // workspace layout (bytes)
  char* ws = (char*)d_ws;
  const size_t SZ_X = (size_t)BT_ * D_ * 2;   // 8 MB
  const size_t SZ_W = (size_t)D_ * D_ * 2;    // 2 MB
  __bf16* xb   = (__bf16*)(ws);
  __bf16* wqb  = (__bf16*)(ws + SZ_X);
  __bf16* wkb  = (__bf16*)(ws + SZ_X + 1 * SZ_W);
  __bf16* wvb  = (__bf16*)(ws + SZ_X + 2 * SZ_W);
  __bf16* wob  = (__bf16*)(ws + SZ_X + 3 * SZ_W);
  __bf16* Qb   = (__bf16*)(ws + SZ_X + 4 * SZ_W);              // [B,H,T,DK]
  __bf16* Kb   = (__bf16*)(ws + 2 * SZ_X + 4 * SZ_W);          // [B,H,T,DK]
  __bf16* Vtb  = (__bf16*)(ws + 3 * SZ_X + 4 * SZ_W);          // [B,H,DK,T]
  __bf16* ctxb = (__bf16*)(ws + 4 * SZ_X + 4 * SZ_W);          // [B,T,D]

  const int nx = BT_ * D_;   // 4194304
  const int nw = D_ * D_;    // 1048576
  cvt_f32_bf16_kernel<<<nx / (256 * 4), 256, 0, stream>>>(x,  xb,  nx);
  cvt_f32_bf16_kernel<<<nw / (256 * 4), 256, 0, stream>>>(wq, wqb, nw);
  cvt_f32_bf16_kernel<<<nw / (256 * 4), 256, 0, stream>>>(wk, wkb, nw);
  cvt_f32_bf16_kernel<<<nw / (256 * 4), 256, 0, stream>>>(wv, wvb, nw);
  cvt_f32_bf16_kernel<<<nw / (256 * 4), 256, 0, stream>>>(wo, wob, nw);

  // 3 proj * 256 row tiles * 16 heads = 12288 waves, 8 waves/block
  qkv_rope_kernel<<<12288 / 8, 256, 0, stream>>>(xb, wqb, wkb, wvb, bq, bk, bv,
                                                 Qb, Kb, Vtb);
  // B*H blocks x 16 query-tile groups, 8 waves each
  flash_attn_kernel<<<B_ * H_ * 16, 256, 0, stream>>>(Qb, Kb, Vtb, ctxb);
  // 256 * 16 = 4096 waves
  out_proj_kernel<<<4096 / 8, 256, 0, stream>>>(ctxb, wob, bo, out);
}
